// PointerNet_67190468378826
// MI455X (gfx1250) — matrix operand order repaired and checked by reading, not compiled, not run on hardware
//
#include <hip/hip_runtime.h>
#include <hip/hip_bf16.h>

typedef __attribute__((ext_vector_type(16))) __bf16 v16bf;
typedef __attribute__((ext_vector_type(8)))  __bf16 v8bf;
typedef __attribute__((ext_vector_type(8)))  float  v8f;

// Types matching the async-LDS builtin prototype (from hipcc diagnostic):
// param = "__attribute__((__vector_size__(4 * sizeof(int)))) int *"
typedef int v4i_gnu __attribute__((vector_size(16)));
typedef __attribute__((address_space(1))) v4i_gnu gv4i;   // global
typedef __attribute__((address_space(3))) v4i_gnu lv4i;   // LDS

#define NUM_HEADS 4
#define B_DIM 16
#define S_DIM 1024
#define T_DIM 512
#define Q_DIM 256
#define E_DIM 512
#define NEG_INF (-1e9f)

static __device__ __forceinline__ __bf16 f2bf(float f) { return (__bf16)f; }
static __device__ __forceinline__ float lrelu(float x) { return x >= 0.f ? x : 0.01f * x; }

// ---- CDNA5 async global->LDS (ASYNCcnt path)
static __device__ __forceinline__ void async_copy_b128(const void* gsrc, void* lds_dst) {
#if __has_builtin(__builtin_amdgcn_global_load_async_to_lds_b128)
    __builtin_amdgcn_global_load_async_to_lds_b128(
        (gv4i*)gsrc, (lv4i*)lds_dst, 0, 0);
#else
    unsigned lds32 = (unsigned)(unsigned long long)
        ((__attribute__((address_space(3))) char*)lds_dst);
    asm volatile("global_load_async_to_lds_b128 %0, %1, off"
                 :: "v"(lds32), "v"(gsrc) : "memory");
#endif
}

#if __has_builtin(__builtin_amdgcn_s_wait_asynccnt)
#define WAIT_ASYNC(n) __builtin_amdgcn_s_wait_asynccnt(n)
#else
#define WAIT_ASYNC(n) asm volatile("s_wait_asynccnt " #n ::: "memory")
#endif

// Assemble a 16-element bf16 A/B fragment from two 8-element contiguous chunks.
static __device__ __forceinline__ v16bf frag16(const __bf16* lo, const __bf16* hi) {
    v8bf l = *(const v8bf*)lo;
    v8bf h = *(const v8bf*)hi;
    v16bf r;
#pragma unroll
    for (int i = 0; i < 8; ++i) { r[i] = l[i]; r[i + 8] = h[i]; }
    return r;
}

// ---------------------------------------------------------------------------
// Stage 0a: Enc f32 -> bf16 (row-major (16384 x 512)), packed 16B stores.
// ---------------------------------------------------------------------------
__global__ __launch_bounds__(256)
void cvt_enc_bf16(const float4* __restrict__ in, v8bf* __restrict__ out) {
    size_t i = (size_t)blockIdx.x * 256 + threadIdx.x;   // one 8-float chunk
    float4 a = in[2 * i], b = in[2 * i + 1];
    v8bf o;
    o[0] = f2bf(a.x); o[1] = f2bf(a.y); o[2] = f2bf(a.z); o[3] = f2bf(a.w);
    o[4] = f2bf(b.x); o[5] = f2bf(b.y); o[6] = f2bf(b.z); o[7] = f2bf(b.w);
    out[i] = o;
}

// ---------------------------------------------------------------------------
// Stage 0b: W_src (512 k x 1024 col) f32 -> Wt (1024 col x 512 k) bf16.
// Tiled transpose so stage-1 B chunks are contiguous along K.
// ---------------------------------------------------------------------------
__global__ __launch_bounds__(256)
void transpose_w_bf16(const float* __restrict__ w, __bf16* __restrict__ wt) {
    __shared__ float tile[32][33];
    const int tx = threadIdx.x, ty = threadIdx.y;        // 32 x 8
    const int col0 = blockIdx.x * 32, k0 = blockIdx.y * 32;
#pragma unroll
    for (int j = 0; j < 4; ++j)
        tile[ty + j * 8][tx] = w[(size_t)(k0 + ty + j * 8) * (NUM_HEADS * Q_DIM) + col0 + tx];
    __syncthreads();
#pragma unroll
    for (int j = 0; j < 4; ++j)
        wt[(size_t)(col0 + ty + j * 8) * E_DIM + k0 + tx] = f2bf(tile[tx][ty + j * 8]);
}

// ---------------------------------------------------------------------------
// Stage 1: C[m, q] = sum_h w_comb[h] * leaky_relu( (Enc @ W_src)[m, h*Q + q] )
// bf16 WMMA, LDS tiles filled by double-buffered async global->LDS copies.
// Block: 256 thr (8 waves), tile 128 m x 16 combined q; 64 WMMA per wave.
// ---------------------------------------------------------------------------
__global__ __launch_bounds__(256)
void pointer_net_proj_combine(const __bf16* __restrict__ encbf,
                              const __bf16* __restrict__ wt,
                              const float*  __restrict__ wcomb,
                              __bf16* __restrict__ cbuf) {
    const int tid  = threadIdx.x;
    const int lane = tid & 31;
    const int wave = tid >> 5;
    const int mBase = blockIdx.x * 128;
    const int qBase = blockIdx.y * 16;

    __shared__ __bf16 As[2][128][40];            // row = 80B (16B aligned chunks)
    __shared__ __bf16 Bs[2][NUM_HEADS][16][40];  // [head][n][k]

    // Per-thread async issue for one K-chunk: 2 A chunks + 1 B chunk (16B each).
    const int ca0 = tid, ca1 = tid + 256;
    const int am0 = ca0 >> 2, ak0 = (ca0 & 3) * 8;
    const int am1 = ca1 >> 2, ak1 = (ca1 & 3) * 8;
    const int bc = tid >> 2, bk = (tid & 3) * 8;
    const int bh = bc >> 4, bn = bc & 15;
    const __bf16* aG0 = encbf + (size_t)(mBase + am0) * E_DIM + ak0;
    const __bf16* aG1 = encbf + (size_t)(mBase + am1) * E_DIM + ak1;
    const __bf16* bG  = wt + (size_t)(bh * Q_DIM + qBase + bn) * E_DIM + bk;

#define ISSUE_CHUNK(kc, bufi)                                         \
    do {                                                              \
        async_copy_b128(aG0 + (kc), &As[bufi][am0][ak0]);             \
        async_copy_b128(aG1 + (kc), &As[bufi][am1][ak1]);             \
        async_copy_b128(bG  + (kc), &Bs[bufi][bh][bn][bk]);           \
    } while (0)

    v8f acc[NUM_HEADS];
#pragma unroll
    for (int h = 0; h < NUM_HEADS; ++h)
#pragma unroll
        for (int i = 0; i < 8; ++i) acc[h][i] = 0.f;

    const int m16 = lane & 15;
    const int k0a = (lane < 16) ? 0 : 16;   // A: K = k0a + e
    const int k0b = (lane < 16) ? 0 : 16;   // B: K = k0b + e
    const int row = wave * 16 + m16;

    ISSUE_CHUNK(0, 0);
    for (int i = 0; i < E_DIM / 32; ++i) {
        const int bufi = i & 1;
        if (i < E_DIM / 32 - 1) {
            ISSUE_CHUNK((i + 1) * 32, (i + 1) & 1);
            WAIT_ASYNC(3);                   // chunk i (oldest 3) has landed
        } else {
            WAIT_ASYNC(0);
        }
        __syncthreads();                     // all threads' chunk-i data visible

        v16bf afrag = frag16(&As[bufi][row][k0a], &As[bufi][row][k0a + 8]);
#pragma unroll
        for (int h = 0; h < NUM_HEADS; ++h) {
            v16bf bfrag = frag16(&Bs[bufi][h][m16][k0b], &Bs[bufi][h][m16][k0b + 8]);
            acc[h] = __builtin_amdgcn_wmma_f32_16x16x32_bf16(
                false, afrag, false, bfrag, (short)0, acc[h], false, false);
        }
        __syncthreads();                     // done reading buf before it is refilled
    }
#undef ISSUE_CHUNK

    const float w0 = wcomb[0], w1 = wcomb[1], w2 = wcomb[2], w3 = wcomb[3];
    const int q = qBase + m16;
#pragma unroll
    for (int v = 0; v < 8; ++v) {
        float c = w0 * lrelu(acc[0][v]) + w1 * lrelu(acc[1][v])
                + w2 * lrelu(acc[2][v]) + w3 * lrelu(acc[3][v]);
        int m = mBase + wave * 16 + ((lane >> 4) << 3) + v;   // D: m=(lane/16)*8+v
        cbuf[(size_t)m * Q_DIM + q] = f2bf(c);
    }
}

// ---------------------------------------------------------------------------
// Stage 2: logits[t,b,s] = query[t,b,:] . C[b,s,:]  (masked -> NEG_INF*sum(w)),
// fused softmax over s, all in registers. Block = 16 t-rows x S=1024, one b.
// ---------------------------------------------------------------------------
__global__ __launch_bounds__(256)
void pointer_net_attn_softmax(const float* __restrict__ qvec,
                              const int*   __restrict__ mask,
                              const float* __restrict__ wcomb,
                              const __bf16* __restrict__ cbuf,
                              float* __restrict__ out) {
    const int tid  = threadIdx.x;
    const int lane = tid & 31;
    const int wave = tid >> 5;
    const int tBase = blockIdx.x * 16;
    const int b     = blockIdx.y;

    __shared__ __bf16 Aq[16][264];     // 256 + pad(8): row = 528B, 16B aligned
    __shared__ float  redbuf[16][8];
    __shared__ float  rowstat[16];

#pragma unroll
    for (int i = 0; i < 16; ++i)
        Aq[i][tid] = f2bf(qvec[((size_t)(tBase + i) * B_DIM + b) * Q_DIM + tid]);
    __syncthreads();

    v8f acc[8];
#pragma unroll
    for (int t8 = 0; t8 < 8; ++t8)
#pragma unroll
        for (int i = 0; i < 8; ++i) acc[t8][i] = 0.f;

    const int m16 = lane & 15;
    const int k0a = (lane < 16) ? 0 : 16;
    const int k0b = (lane < 16) ? 0 : 16;
    const int sCol = wave * 128;

    for (int kc = 0; kc < Q_DIM; kc += 32) {
        v16bf afrag = frag16(&Aq[m16][kc + k0a], &Aq[m16][kc + k0a + 8]);
#pragma unroll
        for (int t8 = 0; t8 < 8; ++t8) {
            int s = sCol + t8 * 16 + m16;
            const __bf16* bp = cbuf + ((size_t)(b * S_DIM + s)) * Q_DIM + kc + k0b;
            v16bf bfrag = frag16(bp, bp + 8);   // 32B contiguous per lane
            acc[t8] = __builtin_amdgcn_wmma_f32_16x16x32_bf16(
                false, afrag, false, bfrag, (short)0, acc[t8], false, false);
        }
    }

    const float wsum = wcomb[0] + wcomb[1] + wcomb[2] + wcomb[3];
    const float maskedVal = NEG_INF * wsum;
#pragma unroll
    for (int t8 = 0; t8 < 8; ++t8) {
        int s = sCol + t8 * 16 + m16;
        if (mask[b * S_DIM + s] != 0) {
#pragma unroll
            for (int v = 0; v < 8; ++v) acc[t8][v] = maskedVal;
        }
    }

    const int rBase = (lane >> 4) << 3;        // D rows: r = rBase + v

    // ---- row max
    float pm[8];
#pragma unroll
    for (int v = 0; v < 8; ++v) {
        float m = acc[0][v];
#pragma unroll
        for (int t8 = 1; t8 < 8; ++t8) m = fmaxf(m, acc[t8][v]);
#pragma unroll
        for (int off = 1; off < 16; off <<= 1) m = fmaxf(m, __shfl_xor(m, off, 16));
        pm[v] = m;
    }
    if ((lane & 15) == 0)
#pragma unroll
        for (int v = 0; v < 8; ++v) redbuf[rBase + v][wave] = pm[v];
    __syncthreads();
    if (tid < 16) {
        float m = redbuf[tid][0];
#pragma unroll
        for (int w = 1; w < 8; ++w) m = fmaxf(m, redbuf[tid][w]);
        rowstat[tid] = m;
    }
    __syncthreads();

    // ---- exp + row sum
    float rmax[8];
#pragma unroll
    for (int v = 0; v < 8; ++v) rmax[v] = rowstat[rBase + v];
    float ps[8];
#pragma unroll
    for (int v = 0; v < 8; ++v) ps[v] = 0.f;
#pragma unroll
    for (int t8 = 0; t8 < 8; ++t8)
#pragma unroll
        for (int v = 0; v < 8; ++v) {
            float e = __expf(acc[t8][v] - rmax[v]);
            acc[t8][v] = e;
            ps[v] += e;
        }
#pragma unroll
    for (int v = 0; v < 8; ++v)
#pragma unroll
        for (int off = 1; off < 16; off <<= 1) ps[v] += __shfl_xor(ps[v], off, 16);
    if ((lane & 15) == 0)
#pragma unroll
        for (int v = 0; v < 8; ++v) redbuf[rBase + v][wave] = ps[v];
    __syncthreads();
    if (tid < 16) {
        float s = 0.f;
#pragma unroll
        for (int w = 0; w < 8; ++w) s += redbuf[tid][w];
        rowstat[tid] = s;
    }
    __syncthreads();

    // ---- normalize + write out[t, b, s]
    float inv[8];
#pragma unroll
    for (int v = 0; v < 8; ++v) inv[v] = 1.f / rowstat[rBase + v];
#pragma unroll
    for (int t8 = 0; t8 < 8; ++t8) {
        int s = sCol + t8 * 16 + m16;
#pragma unroll
        for (int v = 0; v < 8; ++v) {
            int t = tBase + rBase + v;
            out[((size_t)t * B_DIM + b) * S_DIM + s] = acc[t8][v] * inv[v];
        }
    }
}

extern "C" void kernel_launch(void* const* d_in, const int* in_sizes, int n_in,
                              void* d_out, int out_size, void* d_ws, size_t ws_size,
                              hipStream_t stream) {
    (void)in_sizes; (void)n_in; (void)out_size; (void)ws_size;
    const float* enc   = (const float*)d_in[0];  // (B,S,E)
    const int*   mask  = (const int*)  d_in[1];  // (B,S) bool-as-int
    const float* qvec  = (const float*)d_in[2];  // (T,B,Q)
    const float* wsrc  = (const float*)d_in[3];  // (E, H*Q)
    const float* wcomb = (const float*)d_in[4];  // (H,)
    float* out = (float*)d_out;                  // (T,B,S)

    // Workspace layout (bytes): EncBf 16MB | Wt 1MB | C 8MB
    char* ws = (char*)d_ws;
    __bf16* encbf = (__bf16*)ws;                                      // (16384,512)
    __bf16* wt    = (__bf16*)(ws + (size_t)B_DIM * S_DIM * E_DIM * 2); // (1024,512)
    __bf16* cbuf  = (__bf16*)(ws + (size_t)B_DIM * S_DIM * E_DIM * 2
                                 + (size_t)NUM_HEADS * Q_DIM * E_DIM * 2); // (16384,256)

    // Stage 0: one-shot precision conversion / transpose.
    size_t encElems = (size_t)B_DIM * S_DIM * E_DIM;
    cvt_enc_bf16<<<dim3((unsigned)(encElems / 8 / 256)), 256, 0, stream>>>(
        (const float4*)enc, (v8bf*)encbf);
    transpose_w_bf16<<<dim3((NUM_HEADS * Q_DIM) / 32, E_DIM / 32),
                       dim3(32, 8), 0, stream>>>(wsrc, wt);

    // Stage 1: projection GEMM + leaky_relu + head combine.
    dim3 g1((B_DIM * S_DIM) / 128, Q_DIM / 16);  // (128, 16)
    pointer_net_proj_combine<<<g1, 256, 0, stream>>>(encbf, wt, wcomb, cbuf);

    // Stage 2: attention scores + fused masked softmax.
    dim3 g2(T_DIM / 16, B_DIM);                  // (32, 16)
    pointer_net_attn_softmax<<<g2, 256, 0, stream>>>(qvec, mask, wcomb, cbuf, out);
}
